// ScaledSelfAttentionHead_25881472926640
// MI455X (gfx1250) — matrix operand order repaired and checked by reading, not compiled, not run on hardware
//
#include <hip/hip_runtime.h>
#include <hip/hip_bf16.h>

// ---------------------------------------------------------------------------
// Scaled self-attention head, column-softmax variant (softmax over query axis)
// B=16, T=2048, D=1024, H=128, fp32 in/out, bf16 WMMA internally.
// Async global->LDS DMA (ASYNCcnt) double-buffered against WMMA compute.
// ---------------------------------------------------------------------------

#define B_ 16
#define T_ 2048
#define D_ 1024
#define H_ 128
#define NEG_BIG (-3.0e38f)

typedef __attribute__((ext_vector_type(16))) __bf16 v16bf;
typedef __attribute__((ext_vector_type(8)))  float  v8f;
typedef __attribute__((ext_vector_type(4)))  int    v4i;

#if __has_builtin(__builtin_amdgcn_global_load_async_to_lds_b128) && \
    __has_builtin(__builtin_amdgcn_s_wait_asynccnt)
#define HAVE_ASYNC_LDS 1
typedef __attribute__((address_space(1))) v4i* gv4i_p;  // global int4*
typedef __attribute__((address_space(3))) v4i* lv4i_p;  // LDS int4*
#else
#define HAVE_ASYNC_LDS 0
#endif

// Wait until at most N async (global<->LDS) ops remain outstanding for this
// wave. Async loads complete in order, so N=16 == "previous tile done".
template <int N>
__device__ __forceinline__ void async_wait() {
#if HAVE_ASYNC_LDS
  __builtin_amdgcn_s_wait_asynccnt(N);
#endif
}

// --- WMMA wrapper: D = A(16x32 bf16) * B(32x16 bf16) + C(16x16 f32) ---------
__device__ __forceinline__ v8f wmma_bf16(v16bf a, v16bf b, v8f c) {
  return __builtin_amdgcn_wmma_f32_16x16x32_bf16(
      /*neg_a=*/false, a, /*neg_b=*/false, b,
      /*c_mod=*/(short)0, c, /*reuse_a=*/false, /*reuse_b=*/false);
}

// --- A-fragment (16x32, bf16, row-major src with row stride lda elements) ---
__device__ __forceinline__ v16bf load_a_frag(const __bf16* at, int lda,
                                             int m0, int k0, int lane) {
  const int half = lane >> 4, m = lane & 15;
  const __bf16* p = at + (size_t)(m0 + m) * lda + k0 + half * 8;
  union { v16bf v; uint4 q[2]; } u;
  u.q[0] = *reinterpret_cast<const uint4*>(p);
  u.q[1] = *reinterpret_cast<const uint4*>(p + 16);
  return u.v;
}

// Same A layout, but source is f32 (converted on the fly) — used for x.
__device__ __forceinline__ v16bf load_a_frag_f32(const float* at, int lda,
                                                 int m0, int k0, int lane) {
  const int half = lane >> 4, m = lane & 15;
  const float* p = at + (size_t)(m0 + m) * lda + k0 + half * 8;
  v16bf f;
#pragma unroll
  for (int i = 0; i < 8; ++i) f[i] = (__bf16)p[i];
#pragma unroll
  for (int i = 0; i < 8; ++i) f[8 + i] = (__bf16)p[16 + i];
  return f;
}

// --- B-fragment (32x16 K x N, bf16). Row n of src holds column n of B. ------
__device__ __forceinline__ v16bf load_b_frag(const __bf16* bt, int ldb,
                                             int n0, int k0, int lane) {
  const int half = lane >> 4, n = lane & 15;
  const __bf16* p = bt + (size_t)(n0 + n) * ldb + k0 + half * 8;
  union { v16bf v; uint4 q[2]; } u;
  u.q[0] = *reinterpret_cast<const uint4*>(p);
  u.q[1] = *reinterpret_cast<const uint4*>(p + 16);
  return u.v;
}

// --- Cooperative tile staging: 128 rows x (CPR*8) bf16 elems into LDS. ------
// CPR = 16B chunks per row. 256 threads; CPR/2 async ops per thread
// (== CPR/2 async instructions per wave).
template <int CPR>
__device__ __forceinline__ void stage_tile(const __bf16* __restrict__ gsrc,
                                           int gstride, __bf16* lds, int tid) {
#pragma unroll
  for (int chunk = tid; chunk < 128 * CPR; chunk += 256) {
    const int row = chunk / CPR;
    const int off = (chunk % CPR) * 8;
    const __bf16* g = gsrc + (size_t)row * gstride + off;
    __bf16* l = lds + row * (CPR * 8) + off;
#if HAVE_ASYNC_LDS
    __builtin_amdgcn_global_load_async_to_lds_b128((gv4i_p)g, (lv4i_p)l, 0, 0);
#else
    *reinterpret_cast<uint4*>(l) = *reinterpret_cast<const uint4*>(g);
#endif
  }
}

// ---------------------------------------------------------------------------
// Kernel 0: pack Wq|Wk|Wv (each [D,H] f32) into Wt [3][H][D] bf16 (transposed)
// ---------------------------------------------------------------------------
__global__ void pack_w_kernel(const float* __restrict__ Wq,
                              const float* __restrict__ Wk,
                              const float* __restrict__ Wv,
                              __bf16* __restrict__ Wt) {
  int idx = blockIdx.x * blockDim.x + threadIdx.x;
  if (idx >= 3 * H_ * D_) return;
  int mat = idx / (H_ * D_);
  int rem = idx % (H_ * D_);
  int h = rem / D_;
  int d = rem % D_;
  const float* W = (mat == 0) ? Wq : (mat == 1) ? Wk : Wv;
  Wt[idx] = (__bf16)W[(size_t)d * H_ + h];
}

// ---------------------------------------------------------------------------
// Kernel 1: fused QKV projection. Block = 128 rows of x. 8 waves x 16 rows.
// 12 weight chunks (3 mats x 4 K-chunks of 128h x 256k bf16 = 64KB) streamed
// through a double-buffered 128KB LDS pipeline: prefetch chunk c+1 while
// WMMAs consume chunk c. q gets *H^-0.5 baked in; v stored transposed.
// ---------------------------------------------------------------------------
__global__ __launch_bounds__(256) void qkv_kernel(
    const float* __restrict__ x, const __bf16* __restrict__ Wt,
    const float* __restrict__ bq, const float* __restrict__ bk,
    const float* __restrict__ bv, __bf16* __restrict__ qs,
    __bf16* __restrict__ kb, __bf16* __restrict__ vt) {
  __shared__ __bf16 wlds[2][128 * 256];  // 2 x 64KB
  const int tid = threadIdx.x;
  const int lane = tid & 31;
  const int wave = tid >> 5;
  const int rowBase = blockIdx.x * 128 + wave * 16;  // row in [0, B*T)
  const int b = rowBase / T_;
  const int tBase = rowBase % T_;
  const float scale = 0.08838834764831845f;  // 128^-0.5
  const int mrow = (lane >> 4) * 8;

  // chunk c: matrix c>>2, K range [(c&3)*256, +256)
  stage_tile<32>(Wt + (size_t)(0 >> 2) * H_ * D_ + (0 & 3) * 256, D_,
                 wlds[0], tid);

  v8f acc[8] = {};
  for (int c = 0; c < 12; ++c) {
    const int kc = c & 3;
    __bf16* cur = wlds[c & 1];
    if (c + 1 < 12) {
      const __bf16* wnext =
          Wt + (size_t)((c + 1) >> 2) * H_ * D_ + ((c + 1) & 3) * 256;
      stage_tile<32>(wnext, D_, wlds[(c + 1) & 1], tid);
      async_wait<16>();  // chunk c done; chunk c+1 in flight
    } else {
      async_wait<0>();
    }
    __syncthreads();

#pragma unroll
    for (int ki = 0; ki < 8; ++ki) {  // 8 K-steps of 32
      v16bf a = load_a_frag_f32(x, D_, rowBase, kc * 256 + ki * 32, lane);
#pragma unroll
      for (int nt = 0; nt < 8; ++nt) {
        v16bf bf = load_b_frag(cur, 256, nt * 16, ki * 32, lane);
        acc[nt] = wmma_bf16(a, bf, acc[nt]);
      }
    }

    if (kc == 3) {  // matrix complete: bias + store, then reset accumulators
      const int mat = c >> 2;
      if (mat == 0) {
#pragma unroll
        for (int nt = 0; nt < 8; ++nt) {
          const int col = nt * 16 + (lane & 15);
          const float bb = bq[col];
#pragma unroll
          for (int r = 0; r < 8; ++r)
            qs[(size_t)(rowBase + mrow + r) * H_ + col] =
                (__bf16)((acc[nt][r] + bb) * scale);
        }
      } else if (mat == 1) {
#pragma unroll
        for (int nt = 0; nt < 8; ++nt) {
          const int col = nt * 16 + (lane & 15);
          const float bb = bk[col];
#pragma unroll
          for (int r = 0; r < 8; ++r)
            kb[(size_t)(rowBase + mrow + r) * H_ + col] =
                (__bf16)(acc[nt][r] + bb);
        }
      } else {
        // v transposed: lane owns column h, 8 contiguous t -> one b128 store.
#pragma unroll
        for (int nt = 0; nt < 8; ++nt) {
          const int col = nt * 16 + (lane & 15);
          const float bb = bv[col];
          union { __bf16 h[8]; uint4 q; } pk;
#pragma unroll
          for (int r = 0; r < 8; ++r) pk.h[r] = (__bf16)(acc[nt][r] + bb);
          *reinterpret_cast<uint4*>(
              &vt[((size_t)b * H_ + col) * T_ + tBase + mrow]) = pk.q;
        }
      }
#pragma unroll
      for (int nt = 0; nt < 8; ++nt) acc[nt] = (v8f){};
    }
    __syncthreads();  // all waves done reading `cur` before it is re-staged
  }
}

// ---------------------------------------------------------------------------
// Kernel 2: per-key-column softmax stats over the query axis.
// Block = (batch, 128-key block). K tile staged once in LDS, then loop
// q-blocks from the diagonal; online per-column (max,sum); waves combine.
// ---------------------------------------------------------------------------
__global__ __launch_bounds__(256) void colstats_kernel(
    const __bf16* __restrict__ qs, const __bf16* __restrict__ kbuf,
    float* __restrict__ colm, float* __restrict__ colz) {
  __shared__ __bf16 klds[128 * 128];  // 32KB
  __shared__ float mLDS[8][128];
  __shared__ float zLDS[8][128];
  const int tid = threadIdx.x;
  const int lane = tid & 31;
  const int wave = tid >> 5;
  const int b = blockIdx.x >> 4;
  const int kblk = blockIdx.x & 15;
  const int kBase = kblk * 128;
  const __bf16* qb = qs + (size_t)b * T_ * H_;
  const __bf16* kbp = kbuf + (size_t)b * T_ * H_;

  stage_tile<16>(kbp + (size_t)kBase * H_, H_, klds, tid);
  async_wait<0>();
  __syncthreads();

  float mrun[8], zrun[8];
#pragma unroll
  for (int nt = 0; nt < 8; ++nt) { mrun[nt] = NEG_BIG; zrun[nt] = 0.f; }

  for (int qblk = kblk; qblk < T_ / 128; ++qblk) {
    const int qRow = qblk * 128 + wave * 16;
    v16bf afr[4];
#pragma unroll
    for (int ks = 0; ks < 4; ++ks)
      afr[ks] = load_a_frag(qb, H_, qRow, ks * 32, lane);
#pragma unroll
    for (int nt = 0; nt < 8; ++nt) {
      v8f acc = {};
#pragma unroll
      for (int ks = 0; ks < 4; ++ks) {
        v16bf bf = load_b_frag(klds, 128, nt * 16, ks * 32, lane);
        acc = wmma_bf16(afr[ks], bf, acc);
      }
      const int col = kBase + nt * 16 + (lane & 15);
      const int rowOff = qRow + ((lane >> 4) ? 8 : 0);
      float s[8], tmax = NEG_BIG;
      bool keep[8];
#pragma unroll
      for (int r = 0; r < 8; ++r) {
        keep[r] = (rowOff + r) >= col;  // causal: q >= k
        s[r] = keep[r] ? acc[r] : NEG_BIG;
        tmax = fmaxf(tmax, s[r]);
      }
      tmax = fmaxf(tmax, __shfl_xor(tmax, 16));
      const float newm = fmaxf(mrun[nt], tmax);
      float tsum = 0.f;
#pragma unroll
      for (int r = 0; r < 8; ++r)
        tsum += keep[r] ? __expf(s[r] - newm) : 0.f;
      tsum += __shfl_xor(tsum, 16);
      zrun[nt] = zrun[nt] * __expf(mrun[nt] - newm) + tsum;
      mrun[nt] = newm;
    }
  }
  if (lane < 16) {
#pragma unroll
    for (int nt = 0; nt < 8; ++nt) {
      mLDS[wave][nt * 16 + lane] = mrun[nt];
      zLDS[wave][nt * 16 + lane] = zrun[nt];
    }
  }
  __syncthreads();
  if (tid < 128) {
    float m = NEG_BIG;
#pragma unroll
    for (int w = 0; w < 8; ++w) m = fmaxf(m, mLDS[w][tid]);
    float z = 0.f;
#pragma unroll
    for (int w = 0; w < 8; ++w) z += zLDS[w][tid] * __expf(mLDS[w][tid] - m);
    colm[(size_t)b * T_ + kBase + tid] = m;
    colz[(size_t)b * T_ + kBase + tid] = z;
  }
}

// ---------------------------------------------------------------------------
// Kernel 3: out = P @ v with P = exp(S - m[col]) / Z[col], scores recomputed.
// Block = (batch, 128-query block). K/V tiles double-buffered through async
// LDS DMA (prefetch k-block i+1 while computing on i). P staged per-wave
// through LDS (C-layout -> A-layout). 160KB static LDS (gfx1250: 320KB/WGP).
// ---------------------------------------------------------------------------
__global__ __launch_bounds__(256) void attnout_kernel(
    const __bf16* __restrict__ qs, const __bf16* __restrict__ kbuf,
    const __bf16* __restrict__ vt, const float* __restrict__ colm,
    const float* __restrict__ colz, float* __restrict__ out) {
  __shared__ __bf16 klds[2][128 * 128];  // 2 x 32KB
  __shared__ __bf16 vlds[2][128 * 128];  // 2 x 32KB
  __shared__ __bf16 plds[8][16 * 128];   // 32KB (per-wave private slices)
  const int tid = threadIdx.x;
  const int lane = tid & 31;
  const int wave = tid >> 5;
  const int b = blockIdx.x >> 4;
  const int qblk = blockIdx.x & 15;
  const int qRow = qblk * 128 + wave * 16;
  const __bf16* qb = qs + (size_t)b * T_ * H_;
  const __bf16* kbp = kbuf + (size_t)b * T_ * H_;
  const __bf16* vtb = vt + (size_t)b * H_ * T_;
  const float* cm = colm + (size_t)b * T_;
  const float* cz = colz + (size_t)b * T_;
  const int mrow = (lane >> 4) ? 8 : 0;

  v16bf qfr[4];
#pragma unroll
  for (int ks = 0; ks < 4; ++ks)
    qfr[ks] = load_a_frag(qb, H_, qRow, ks * 32, lane);

  v8f accd[8] = {};

  // Prologue: stage k-block 0 (8+8 async ops per wave).
  stage_tile<16>(kbp, H_, klds[0], tid);
  stage_tile<16>(vtb, T_, vlds[0], tid);

  for (int kblk = 0; kblk <= qblk; ++kblk) {
    const int kBase = kblk * 128;
    const int buf = kblk & 1;
    if (kblk + 1 <= qblk) {  // prefetch next k-block into the other buffer
      stage_tile<16>(kbp + (size_t)(kBase + 128) * H_, H_, klds[buf ^ 1], tid);
      stage_tile<16>(vtb + kBase + 128, T_, vlds[buf ^ 1], tid);
      async_wait<16>();  // current tile done; next tile in flight
    } else {
      async_wait<0>();
    }
    __syncthreads();

    // --- scores + normalize -> P (bf16) into this wave's LDS slice ---
#pragma unroll
    for (int nt = 0; nt < 8; ++nt) {
      v8f acc = {};
#pragma unroll
      for (int ks = 0; ks < 4; ++ks) {
        v16bf bf = load_b_frag(klds[buf], 128, nt * 16, ks * 32, lane);
        acc = wmma_bf16(qfr[ks], bf, acc);
      }
      const int cloc = nt * 16 + (lane & 15);
      const int col = kBase + cloc;
      const float mcol = cm[col];
      const float rz = 1.0f / cz[col];  // Z > 0: diagonal always unmasked
#pragma unroll
      for (int r = 0; r < 8; ++r) {
        const bool keep = (qRow + mrow + r) >= col;
        const float p = keep ? __expf(acc[r] - mcol) * rz : 0.f;
        plds[wave][(mrow + r) * 128 + cloc] = (__bf16)p;
      }
    }
    // --- P (16x128) @ v (128x128 via vlds) ---
#pragma unroll
    for (int ks = 0; ks < 4; ++ks) {
      v16bf a = load_a_frag(&plds[wave][0], 128, 0, ks * 32, lane);
#pragma unroll
      for (int ht = 0; ht < 8; ++ht) {
        v16bf bf = load_b_frag(vlds[buf], 128, ht * 16, ks * 32, lane);
        accd[ht] = wmma_bf16(a, bf, accd[ht]);
      }
    }
    __syncthreads();  // all waves done with buf before it is re-staged
  }
  // --- store f32 output [B,T,H] ---
#pragma unroll
  for (int ht = 0; ht < 8; ++ht) {
    const int h = ht * 16 + (lane & 15);
#pragma unroll
    for (int r = 0; r < 8; ++r) {
      const int t = qRow + mrow + r;
      out[((size_t)b * T_ + t) * H_ + h] = accd[ht][r];
    }
  }
}

// ---------------------------------------------------------------------------
extern "C" void kernel_launch(void* const* d_in, const int* in_sizes, int n_in,
                              void* d_out, int out_size, void* d_ws,
                              size_t ws_size, hipStream_t stream) {
  const float* x  = (const float*)d_in[0];
  const float* Wk = (const float*)d_in[1];
  const float* bk = (const float*)d_in[2];
  const float* Wq = (const float*)d_in[3];
  const float* bq = (const float*)d_in[4];
  const float* Wv = (const float*)d_in[5];
  const float* bv = (const float*)d_in[6];
  float* out = (float*)d_out;

  // Workspace partition (~26.2 MB; all 16B-aligned segments).
  char* w = (char*)d_ws;
  __bf16* Wt = (__bf16*)w;                       // [3][H][D]
  __bf16* qs = Wt + (size_t)3 * H_ * D_;         // [B*T][H], scaled
  __bf16* kb = qs + (size_t)B_ * T_ * H_;        // [B*T][H]
  __bf16* vt = kb + (size_t)B_ * T_ * H_;        // [B][H][T]
  float* colm = (float*)(vt + (size_t)B_ * T_ * H_);  // [B][T]
  float* colz = colm + (size_t)B_ * T_;               // [B][T]

  pack_w_kernel<<<(3 * H_ * D_ + 255) / 256, 256, 0, stream>>>(Wq, Wk, Wv, Wt);
  qkv_kernel<<<(B_ * T_) / 128, 256, 0, stream>>>(x, Wt, bq, bk, bv, qs, kb, vt);
  colstats_kernel<<<B_ * 16, 256, 0, stream>>>(qs, kb, colm, colz);
  attnout_kernel<<<B_ * 16, 256, 0, stream>>>(qs, kb, vt, colm, colz, out);
}
